// SpamClassifier_9010841387354
// MI455X (gfx1250) — matrix operand order repaired and checked by reading, not compiled, not run on hardware
//
#include <hip/hip_runtime.h>

typedef __bf16 bf16_t;
typedef __bf16 v8bf  __attribute__((ext_vector_type(8)));
typedef __bf16 v16bf __attribute__((ext_vector_type(16)));
typedef float  v8f   __attribute__((ext_vector_type(8)));
typedef float  v4f   __attribute__((ext_vector_type(4)));

static constexpr int HID  = 1024;   // hidden size
static constexpr int G4   = 4096;   // 4*H gate rows
static constexpr int NB   = 64;     // batch
static constexpr int TT   = 100;    // timesteps
static constexpr int NK   = 9;      // experts
static constexpr int EMB  = 100;
static constexpr int EMBP = 128;    // padded embedding (K-dim mult of 32)
static constexpr int NOUT = 2;

__device__ __forceinline__ float sigf(float x) { return 1.0f / (1.0f + __expf(-x)); }

// ---------------- prep kernels ----------------

// W_hh fp32 [K][4096][1024] -> bf16 (4-wide)
__global__ void k_conv_whh(const float* __restrict__ src, bf16_t* __restrict__ dst, int n4) {
    int i = blockIdx.x * blockDim.x + threadIdx.x;
    if (i >= n4) return;
    v4f v = ((const v4f*)src)[i];
    bf16_t* d = dst + (size_t)i * 4;
    d[0] = (bf16_t)v.x; d[1] = (bf16_t)v.y; d[2] = (bf16_t)v.z; d[3] = (bf16_t)v.w;
}

// W_ih fp32 [K*4096][100] -> bf16 [K*4096][128] zero-padded
__global__ void k_conv_wih(const float* __restrict__ src, bf16_t* __restrict__ dst, int n) {
    int i = blockIdx.x * blockDim.x + threadIdx.x;
    if (i >= n) return;
    int e = i & (EMBP - 1);
    int row = i >> 7;
    float v = (e < EMB) ? src[(size_t)row * EMB + e] : 0.0f;
    dst[i] = (bf16_t)v;
}

// bias = b_ih + b_hh  [K*4096]
__global__ void k_bias(const float* __restrict__ bih, const float* __restrict__ bhh,
                       float* __restrict__ bias, int n) {
    int i = blockIdx.x * blockDim.x + threadIdx.x;
    if (i < n) bias[i] = bih[i] + bhh[i];
}

// per-timestep routing: route[t][k][p]=b, pos[t][b]=p, cnt[t][k]
__global__ void k_route(const int* __restrict__ tags, int* __restrict__ route,
                        int* __restrict__ posArr, int* __restrict__ cntArr) {
    int t = blockIdx.x;
    if (threadIdx.x != 0) return;
    int cnt[NK];
#pragma unroll
    for (int k = 0; k < NK; ++k) cnt[k] = 0;
    for (int b = 0; b < NB; ++b) {
        int k = tags[b * TT + t];
        int p = cnt[k]++;
        route[(t * NK + k) * NB + p] = b;
        posArr[t * NB + b] = p;
    }
    for (int k = 0; k < NK; ++k) cntArr[t * NK + k] = cnt[k];
}

// gathered embedding panels: xB[t][k][col][128] (column-major K panels for WMMA B)
__global__ void k_xpanel(const int* __restrict__ x, const int* __restrict__ tags,
                         const int* __restrict__ posArr, const float* __restrict__ emb,
                         bf16_t* __restrict__ xB, int n) {
    int i = blockIdx.x * blockDim.x + threadIdx.x;
    if (i >= n) return;
    int e  = i & (EMBP - 1);
    int rb = i >> 7;
    int b  = rb % NB;
    int t  = rb / NB;
    int k  = tags[b * TT + t];
    int p  = posArr[t * NB + b];
    float v = (e < EMB) ? emb[(size_t)x[b * TT + t] * EMB + e] : 0.0f;
    xB[(((size_t)t * NK + k) * NB + p) * EMBP + e] = (bf16_t)v;
}

// init c,h fp32 state and the t=0 gathered-h panel
__global__ void k_init(const float* __restrict__ h0, const float* __restrict__ c0,
                       const int* __restrict__ tags, const int* __restrict__ posArr,
                       float* __restrict__ hbuf, float* __restrict__ cbuf,
                       bf16_t* __restrict__ hB0, int n) {
    int i = blockIdx.x * blockDim.x + threadIdx.x;
    if (i >= n) return;
    int j = i & (HID - 1);
    int b = i >> 10;
    float hv = h0[i];
    hbuf[i] = hv;
    cbuf[i] = c0[i];
    int k0 = tags[b * TT + 0];
    int p0 = posArr[b];
    hB0[((size_t)k0 * NB + p0) * HID + j] = (bf16_t)hv;
}

// ---------------- recurrent step (WMMA) ----------------

// one panel GEMM: acc[nt] += A(16 x kdim) * B(kdim x 16) for NTL column tiles.
// A fragment: lane<16 row=l15, K kk+0..15;
//             lane>=16 row=l15, K kk+16..31.
// B panels are column-major: lane<16 col=nt*16+l15 K kk+0..15; lane>=16 K kk+16..31.
template <int NTL>
__device__ __forceinline__ void gemm_panel(const bf16_t* __restrict__ Arow,
                                           const bf16_t* __restrict__ Bpanel,
                                           int bstride, int kdim,
                                           int half, int l15, v8f (&acc)[NTL]) {
#pragma unroll 2
    for (int kk = 0; kk < kdim; kk += 32) {
        v16bf a = *(const v16bf*)(Arow + kk + half * 16);
#pragma unroll
        for (int nt = 0; nt < NTL; ++nt) {
            const v16bf* bp =
                (const v16bf*)(Bpanel + (size_t)(nt * 16 + l15) * bstride + kk + half * 16);
            acc[nt] = __builtin_amdgcn_wmma_f32_16x16x32_bf16(false, a, false, *bp,
                                                              (short)0, acc[nt], false, false);
        }
    }
}

template <int NTL>
__device__ __forceinline__ void step_compute(int t, int jt, int k, int wave, int half, int l15,
                                             const bf16_t* __restrict__ whh,
                                             const bf16_t* __restrict__ wih,
                                             const bf16_t* __restrict__ xB,
                                             const bf16_t* __restrict__ hBcur,
                                             float (&gl)[4][16][65]) {
    v8f acc[NTL] = {};
    {
        const bf16_t* Arow =
            whh + ((size_t)k * G4 + (size_t)wave * HID + (size_t)jt * 16 + l15) * HID;
        const bf16_t* Bk = hBcur + (size_t)k * NB * HID;
        gemm_panel<NTL>(Arow, Bk, HID, HID, half, l15, acc);
    }
    {
        const bf16_t* Arow =
            wih + ((size_t)k * G4 + (size_t)wave * HID + (size_t)jt * 16 + l15) * EMBP;
        const bf16_t* Bk = xB + ((size_t)t * NK + k) * NB * EMBP;
        gemm_panel<NTL>(Arow, Bk, EMBP, EMBP, half, l15, acc);
    }
    // D tile spill: VGPR vg -> (M = vg + 8*half, N = l15) per ISA C/D layout
    const int m0 = 8 * half;
#pragma unroll
    for (int nt = 0; nt < NTL; ++nt)
#pragma unroll
        for (int vg = 0; vg < 8; ++vg) gl[wave][m0 + vg][nt * 16 + l15] = acc[nt][vg];
}

// grid: (64 j-tiles, 9 experts), block: 128 = 4 waves (one per gate i/f/g/o)
__global__ __launch_bounds__(128) void k_step(
    int t,
    const bf16_t* __restrict__ whh,    // [K][4096][1024] bf16 row-major
    const bf16_t* __restrict__ wih,    // [K][4096][128]  bf16 row-major
    const float*  __restrict__ bias,   // [K][4096]
    const bf16_t* __restrict__ xB,     // [T][K][64][128] column panels
    const bf16_t* __restrict__ hBcur,  // [K][64][1024]   column panels
    bf16_t*       __restrict__ hBnext, // [K][64][1024]
    const int*    __restrict__ route,  // [T][K][64]
    const int*    __restrict__ posArr, // [T][B]
    const int*    __restrict__ cntArr, // [T][K]
    const int*    __restrict__ tags,   // [B][T]
    float* __restrict__ cbuf, float* __restrict__ hbuf)
{
    const int jt = blockIdx.x;            // hidden-unit tile (16 units)
    const int k  = blockIdx.y;            // expert
    const int cnt = cntArr[t * NK + k];
    if (cnt == 0) return;
    const int NTL = (cnt + 15) >> 4;      // active 16-column tiles (1..4)

    const int wave = threadIdx.x >> 5;    // gate index 0..3
    const int lane = threadIdx.x & 31;
    const int half = lane >> 4;
    const int l15  = lane & 15;

    __shared__ float gl[4][16][65];       // [gate][j][col] (+1 pad vs banks)

    switch (NTL) {
        case 1: step_compute<1>(t, jt, k, wave, half, l15, whh, wih, xB, hBcur, gl); break;
        case 2: step_compute<2>(t, jt, k, wave, half, l15, whh, wih, xB, hBcur, gl); break;
        case 3: step_compute<3>(t, jt, k, wave, half, l15, whh, wih, xB, hBcur, gl); break;
        default: step_compute<4>(t, jt, k, wave, half, l15, whh, wih, xB, hBcur, gl); break;
    }
    __syncthreads();

    // ---- fused LSTM pointwise + scatter into next step's panels ----
    const float* bs = bias + (size_t)k * G4;
    const int* rt = route + (t * NK + k) * NB;
    for (int idx = threadIdx.x; idx < cnt * 16; idx += 128) {
        int col = idx >> 4;
        int j   = idx & 15;
        int b   = rt[col];
        int jg  = jt * 16 + j;
        float iv = gl[0][j][col] + bs[0 * HID + jg];
        float fv = gl[1][j][col] + bs[1 * HID + jg];
        float gv = gl[2][j][col] + bs[2 * HID + jg];
        float ov = gl[3][j][col] + bs[3 * HID + jg];
        float cold = cbuf[b * HID + jg];
        float cn = sigf(fv) * cold + sigf(iv) * tanhf(gv);
        float hn = sigf(ov) * tanhf(cn);
        cbuf[b * HID + jg] = cn;
        hbuf[b * HID + jg] = hn;
        if (t + 1 < TT) {
            int kn = tags[b * TT + t + 1];
            int pn = posArr[(t + 1) * NB + b];
            hBnext[((size_t)kn * NB + pn) * HID + jg] = (bf16_t)hn;
        }
    }
}

// ---------------- output ----------------

// out[b] = sigmoid(h[b] . W_fc[OUT-1] + b_fc[OUT-1])
__global__ void k_fc(const float* __restrict__ hbuf, const float* __restrict__ Wfc,
                     const float* __restrict__ bfc, float* __restrict__ out) {
    __shared__ float red[128];
    int b = blockIdx.x;
    const float* wrow = Wfc + (size_t)(NOUT - 1) * HID;
    float s = 0.0f;
    for (int j = threadIdx.x; j < HID; j += 128) s += hbuf[(size_t)b * HID + j] * wrow[j];
    red[threadIdx.x] = s;
    __syncthreads();
    for (int off = 64; off > 0; off >>= 1) {
        if (threadIdx.x < off) red[threadIdx.x] += red[threadIdx.x + off];
        __syncthreads();
    }
    if (threadIdx.x == 0) out[b] = sigf(red[0] + bfc[NOUT - 1]);
}

// d_out layout: out[64] | h[64*1024] | c[64*1024]
__global__ void k_copyout(const float* __restrict__ hbuf, const float* __restrict__ cbuf,
                          float* __restrict__ dout, int n) {
    int i = blockIdx.x * blockDim.x + threadIdx.x;
    if (i >= n) return;
    dout[NB + i] = hbuf[i];
    dout[NB + n + i] = cbuf[i];
}

// ---------------- launch ----------------

extern "C" void kernel_launch(void* const* d_in, const int* in_sizes, int n_in,
                              void* d_out, int out_size, void* d_ws, size_t ws_size,
                              hipStream_t stream) {
    const int*   x    = (const int*)d_in[0];
    const int*   tags = (const int*)d_in[1];
    const float* h0   = (const float*)d_in[2];
    const float* c0   = (const float*)d_in[3];
    const float* emb  = (const float*)d_in[4];
    const float* Wih  = (const float*)d_in[5];
    const float* Whh  = (const float*)d_in[6];
    const float* bih  = (const float*)d_in[7];
    const float* bhh  = (const float*)d_in[8];
    const float* Wfc  = (const float*)d_in[9];
    const float* bfc  = (const float*)d_in[10];
    float* out = (float*)d_out;

    // workspace carve (aligned 256B)
    char* p = (char*)d_ws;
    auto carve = [&](size_t bytes) { void* r = (void*)p; p += (bytes + 255) & ~(size_t)255; return r; };
    bf16_t* whh_bf = (bf16_t*)carve((size_t)NK * G4 * HID * sizeof(bf16_t));   // 75.5 MB
    bf16_t* wih_bf = (bf16_t*)carve((size_t)NK * G4 * EMBP * sizeof(bf16_t));  // 9.4 MB
    float*  bias   = (float*) carve((size_t)NK * G4 * sizeof(float));
    bf16_t* xB     = (bf16_t*)carve((size_t)TT * NK * NB * EMBP * sizeof(bf16_t)); // 14.7 MB
    bf16_t* hB_A   = (bf16_t*)carve((size_t)NK * NB * HID * sizeof(bf16_t));   // 1.18 MB
    bf16_t* hB_B   = (bf16_t*)carve((size_t)NK * NB * HID * sizeof(bf16_t));
    float*  cbuf   = (float*) carve((size_t)NB * HID * sizeof(float));
    float*  hbuf   = (float*) carve((size_t)NB * HID * sizeof(float));
    int*    route  = (int*)   carve((size_t)TT * NK * NB * sizeof(int));
    int*    posArr = (int*)   carve((size_t)TT * NB * sizeof(int));
    int*    cntArr = (int*)   carve((size_t)TT * NK * sizeof(int));

    // prep
    {
        int n4 = NK * G4 * HID / 4;
        k_conv_whh<<<(n4 + 255) / 256, 256, 0, stream>>>(Whh, whh_bf, n4);
        int ni = NK * G4 * EMBP;
        k_conv_wih<<<(ni + 255) / 256, 256, 0, stream>>>(Wih, wih_bf, ni);
        int nb = NK * G4;
        k_bias<<<(nb + 255) / 256, 256, 0, stream>>>(bih, bhh, bias, nb);
        k_route<<<TT, 32, 0, stream>>>(tags, route, posArr, cntArr);
        int nx = TT * NB * EMBP;
        k_xpanel<<<(nx + 255) / 256, 256, 0, stream>>>(x, tags, posArr, emb, xB, nx);
        int nh = NB * HID;
        k_init<<<(nh + 255) / 256, 256, 0, stream>>>(h0, c0, tags, posArr, hbuf, cbuf, hB_A, nh);
    }

    // sequential recurrence: one launch per timestep (graph replay amortizes)
    for (int t = 0; t < TT; ++t) {
        const bf16_t* cur = (t & 1) ? hB_B : hB_A;
        bf16_t*       nxt = (t & 1) ? hB_A : hB_B;
        dim3 grid(HID / 16, NK);
        k_step<<<grid, 128, 0, stream>>>(t, whh_bf, wih_bf, bias, xB, cur, nxt,
                                         route, posArr, cntArr, tags, cbuf, hbuf);
    }

    // output: out | h | c
    k_fc<<<NB, 128, 0, stream>>>(hbuf, Wfc, bfc, out);
    int nh = NB * HID;
    k_copyout<<<(nh + 255) / 256, 256, 0, stream>>>(hbuf, cbuf, out, nh);
}
